// ChannelAttention_18013092840141
// MI455X (gfx1250) — matrix operand order repaired and checked by reading, not compiled, not run on hardware
//
#include <hip/hip_runtime.h>

// ---------------------------------------------------------------------------
// ChannelAttention (XCiT-style cross-covariance attention) for MI455X gfx1250
// B=4, N=8192, C=512, heads=4, dh=128.  All GEMMs on v_wmma_f32_16x16x32_bf16.
// Round 3: async global->LDS staging (ASYNCcnt) in the q.k^T kernel.
// ---------------------------------------------------------------------------

#define BB      4
#define NN      8192
#define CC      512
#define HEADS   4
#define DH      128
#define C3      (3 * CC)

typedef __attribute__((ext_vector_type(16))) __bf16 v16bf;
typedef __attribute__((ext_vector_type(8)))  __bf16 v8bf;
typedef __attribute__((ext_vector_type(8)))  float  v8f;

union V16U { v16bf v; v8bf h[2]; };

__device__ __forceinline__ unsigned short f2bf(float f) {
    __bf16 b = (__bf16)f;                       // native v_cvt_pk_bf16_f32
    return __builtin_bit_cast(unsigned short, b);
}
__device__ __forceinline__ float bf2f(unsigned short h) {
    return __uint_as_float(((unsigned)h) << 16);
}
__device__ __forceinline__ v8f vzero() {
    v8f z = {0.f, 0.f, 0.f, 0.f, 0.f, 0.f, 0.f, 0.f};
    return z;
}

// Async DMA: LDS[lds_addr + 0..15] = MEM[gaddr + 0..15], tracked by ASYNCcnt.
__device__ __forceinline__ void async_copy_b128(unsigned lds_addr,
                                                const void* gptr) {
    asm volatile("global_load_async_to_lds_b128 %0, %1, off"
                 :: "v"(lds_addr), "v"((unsigned long long)(size_t)gptr)
                 : "memory");
}
__device__ __forceinline__ void wait_async() {
    asm volatile("s_wait_asynccnt 0" ::: "memory");
}

// A fragment (16x32 bf16): lane 0-15 -> M=lane, K 0..7 & 16..23;
//                          lane 16-31 -> M=lane-16, K 8..15 & 24..31.
// `base` is row-major [M][K] (LDS or global), ld in elements.
__device__ __forceinline__ v16bf load_frag_a(const unsigned short* base, int ld,
                                             int m0, int k0) {
    int lane = threadIdx.x & 31;
    int m  = m0 + (lane & 15);
    int kb = k0 + ((lane >> 4) << 3);
    V16U r;
    r.h[0] = *(const v8bf*)(base + m * ld + kb);
    r.h[1] = *(const v8bf*)(base + m * ld + kb + 16);
    return r.v;
}

// B fragment (32x16 bf16): lane 0-15 -> col=lane, K 0..7 & 16..23;
//                          lane 16-31 -> col=lane-16, K 8..15 & 24..31.
// `base` stores B column-index-major: [col][K] contiguous K.
__device__ __forceinline__ v16bf load_frag_b(const unsigned short* base, int ld,
                                             int n0, int k0) {
    int lane = threadIdx.x & 31;
    int n  = n0 + (lane & 15);
    int kb = k0 + ((lane >> 4) << 3);
    V16U r;
    r.h[0] = *(const v8bf*)(base + n * ld + kb);
    r.h[1] = *(const v8bf*)(base + n * ld + kb + 16);
    return r.v;
}

#define WMMA_BF16(A, Bm, Cm) \
    __builtin_amdgcn_wmma_f32_16x16x32_bf16(false, (A), false, (Bm), (short)0, (Cm), false, false)

// ---------------------------------------------------------------------------
// Kernel 0: convert weights fp32 -> bf16 (once per call; ~6 MB traffic)
// ---------------------------------------------------------------------------
__global__ void k_convert_weights(const float* __restrict__ wqkv_f,
                                  const float* __restrict__ wout_f,
                                  unsigned short* __restrict__ wqkv_b,
                                  unsigned short* __restrict__ wout_b) {
    int t = blockIdx.x * blockDim.x + threadIdx.x;
    if (t < C3 * CC)  wqkv_b[t] = f2bf(wqkv_f[t]);
    if (t < CC * CC)  wout_b[t] = f2bf(wout_f[t]);
}

// ---------------------------------------------------------------------------
// Kernel 1: qkv = x @ W_qkv^T, written channel-major bf16 [3][B][H][dh][N].
// Grid (256, 12); block 256 (8 waves).  Tile 128x128, K-step 32,
// double-buffered LDS staging (fp32 -> bf16 convert in flight).
// ---------------------------------------------------------------------------
__global__ void k_qkv_gemm(const float* __restrict__ x,
                           const unsigned short* __restrict__ wqkv_b,
                           unsigned short* __restrict__ qkv) {
    __shared__ unsigned short xtile[2][128 * 40];   // [token][k], ld=40 (pad)
    __shared__ unsigned short wtile[2][128 * 40];   // [ocol ][k], ld=40

    const int tid    = threadIdx.x;
    const int waveId = tid >> 5;
    const int wr = waveId >> 2;          // 0..1  (row block of 64)
    const int wc = waveId & 3;           // 0..3  (col block of 32)
    const int rowBase = blockIdx.x * 128;
    const int colBase = blockIdx.y * 128;

    v8f acc[4][2];
    for (int i = 0; i < 4; ++i)
        for (int j = 0; j < 2; ++j) acc[i][j] = vzero();

    const int r    = tid >> 1;          // staging: row 0..127
    const int half = tid & 1;           // 16-element half of the 32-wide K slab

    auto stage = [&](int buf, int K0) {
        // stage x (fp32 -> bf16)
        const float* src = x + (size_t)(rowBase + r) * CC + K0 + half * 16;
        unsigned short tmp[16];
#pragma unroll
        for (int p = 0; p < 4; ++p) {
            float4 f = *(const float4*)(src + p * 4);
            tmp[p * 4 + 0] = f2bf(f.x); tmp[p * 4 + 1] = f2bf(f.y);
            tmp[p * 4 + 2] = f2bf(f.z); tmp[p * 4 + 3] = f2bf(f.w);
        }
        uint4* d = (uint4*)&xtile[buf][r * 40 + half * 16];
        d[0] = ((const uint4*)tmp)[0];
        d[1] = ((const uint4*)tmp)[1];
        // stage W (bf16 copy)
        const unsigned short* ws =
            wqkv_b + (size_t)(colBase + r) * CC + K0 + half * 16;
        uint4* dw = (uint4*)&wtile[buf][r * 40 + half * 16];
        dw[0] = ((const uint4*)ws)[0];
        dw[1] = ((const uint4*)ws)[1];
    };

    stage(0, 0);
    for (int kk = 0; kk < CC / 32; ++kk) {
        __syncthreads();
        if (kk + 1 < CC / 32) {
            __builtin_prefetch(wqkv_b + (size_t)(colBase + r) * CC +
                               (kk + 1) * 32, 0, 3);
            stage((kk + 1) & 1, (kk + 1) * 32);
        }
        const int buf = kk & 1;
#pragma unroll
        for (int i = 0; i < 4; ++i) {
            v16bf a = load_frag_a(xtile[buf], 40, wr * 64 + i * 16, 0);
#pragma unroll
            for (int j = 0; j < 2; ++j) {
                v16bf b = load_frag_b(wtile[buf], 40, wc * 32 + j * 16, 0);
                acc[i][j] = WMMA_BF16(a, b, acc[i][j]);
            }
        }
    }

    // scatter-store channel-major bf16
    const int lane = tid & 31;
#pragma unroll
    for (int i = 0; i < 4; ++i)
#pragma unroll
        for (int j = 0; j < 2; ++j)
#pragma unroll
            for (int e = 0; e < 8; ++e) {
                int m  = wr * 64 + i * 16 + e + ((lane >> 4) << 3);
                int nc = wc * 32 + j * 16 + (lane & 15);
                int gm = rowBase + m;              // 0..32767
                int b  = gm >> 13;                 // /8192
                int n  = gm & (NN - 1);
                int o  = colBase + nc;             // 0..1535
                int i3 = o >> 9;
                int hh = (o >> 7) & 3;
                int c  = o & 127;
                size_t idx = ((((size_t)i3 * BB + b) * HEADS + hh) * DH + c) *
                                 (size_t)NN + n;
                qkv[idx] = f2bf(acc[i][j][e]);
            }
}

// ---------------------------------------------------------------------------
// Kernel 2: per-channel sum of squares over N for q and k.
// Grid 2*B*H*dh = 4096; block 256.
// ---------------------------------------------------------------------------
__global__ void k_norms(const unsigned short* __restrict__ qkv,
                        float* __restrict__ norms) {
    __shared__ float red[256];
    const int tid = threadIdx.x;
    const int id  = blockIdx.x;            // qk*2048 + bh*128 + c
    const int qk  = id >> 11;
    const int rem = id & 2047;
    const unsigned short* row =
        qkv + ((size_t)qk * (BB * HEADS * DH) + rem) * (size_t)NN;

    float local = 0.f;
#pragma unroll
    for (int it = 0; it < 4; ++it) {
        uint4 raw = *(const uint4*)(row + (size_t)(tid + it * 256) * 8);
        const unsigned short* u = (const unsigned short*)&raw;
#pragma unroll
        for (int e = 0; e < 8; ++e) {
            float f = bf2f(u[e]);
            local += f * f;
        }
    }
    red[tid] = local;
    __syncthreads();
    for (int s = 128; s > 0; s >>= 1) {
        if (tid < s) red[tid] += red[tid + s];
        __syncthreads();
    }
    if (tid == 0) norms[id] = red[0];
}

// ---------------------------------------------------------------------------
// Kernel 3z: zero the fp32 S accumulator (poisoned workspace).
// ---------------------------------------------------------------------------
__global__ void k_zero(float* __restrict__ p, int n) {
    for (int i = blockIdx.x * blockDim.x + threadIdx.x; i < n;
         i += gridDim.x * blockDim.x)
        p[i] = 0.f;
}

// ---------------------------------------------------------------------------
// Kernel 3a: S[b,h] += q[dh, Nslice] * k[dh, Nslice]^T  (K split 16 ways,
// accumulated with global_atomic_add_f32).  Grid (16, 16); block 256.
// Double-buffered staging via GLOBAL_LOAD_ASYNC_TO_LDS_B128 (ASYNCcnt).
// ---------------------------------------------------------------------------
__global__ void k_qk_gemm(const unsigned short* __restrict__ qkv,
                          float* __restrict__ S) {
    __shared__ unsigned short qtile[2][128 * 40];
    __shared__ unsigned short ktile[2][128 * 40];

    const int tid    = threadIdx.x;
    const int waveId = tid >> 5;
    const int lane   = tid & 31;
    const int bh     = blockIdx.x;                 // 0..15
    const int nbase  = blockIdx.y * (NN / 16);     // 512-wide slice

    const unsigned short* qbase =
        qkv + ((size_t)0 * 16 + bh) * (size_t)DH * NN;
    const unsigned short* kbase =
        qkv + ((size_t)1 * 16 + bh) * (size_t)DH * NN;

    v8f acc[8];
#pragma unroll
    for (int j = 0; j < 8; ++j) acc[j] = vzero();

    const int r    = tid >> 1;
    const int half = tid & 1;

    auto stage = [&](int buf, int n0) {
        // each thread DMAs 2x16B of q and 2x16B of k straight into LDS
        unsigned dq = (unsigned)(size_t)&qtile[buf][r * 40 + half * 16];
        unsigned dk = (unsigned)(size_t)&ktile[buf][r * 40 + half * 16];
        const unsigned short* sq = qbase + (size_t)r * NN + n0 + half * 16;
        const unsigned short* sk = kbase + (size_t)r * NN + n0 + half * 16;
        async_copy_b128(dq,      sq);
        async_copy_b128(dq + 16, sq + 8);
        async_copy_b128(dk,      sk);
        async_copy_b128(dk + 16, sk + 8);
    };

    stage(0, nbase);
    const int iters = (NN / 16) / 32;              // 16
    for (int it = 0; it < iters; ++it) {
        wait_async();                               // staged buf is resident
        __syncthreads();
        if (it + 1 < iters) stage((it + 1) & 1, nbase + (it + 1) * 32);
        const int buf = it & 1;
        v16bf a = load_frag_a(qtile[buf], 40, waveId * 16, 0);
#pragma unroll
        for (int j = 0; j < 8; ++j) {
            v16bf b = load_frag_b(ktile[buf], 40, j * 16, 0);
            acc[j] = WMMA_BF16(a, b, acc[j]);
        }
    }

    float* Sbh = S + (size_t)bh * DH * DH;
#pragma unroll
    for (int j = 0; j < 8; ++j)
#pragma unroll
        for (int e = 0; e < 8; ++e) {
            int c = waveId * 16 + e + ((lane >> 4) << 3);
            int d = j * 16 + (lane & 15);
            atomicAdd(&Sbh[c * DH + d], acc[j][e]);
        }
}

// ---------------------------------------------------------------------------
// Kernel 3b: attn = softmax_d( S * temp / (||q_c|| * ||k_d||) ), bf16 out.
// Grid 16; block 256 (8 waves, each wave owns 16 rows).
// ---------------------------------------------------------------------------
__global__ void k_softmax(const float* __restrict__ S,
                          const float* __restrict__ norms,
                          const float* __restrict__ temperature,
                          unsigned short* __restrict__ attn) {
    const int tid    = threadIdx.x;
    const int waveId = tid >> 5;
    const int lane   = tid & 31;
    const int bh     = blockIdx.x;
    const int h      = bh & 3;
    const float temp = temperature[h];
    const float* nq  = norms + bh * DH;
    const float* nk  = norms + 2048 + bh * DH;

    for (int rr = 0; rr < 16; ++rr) {
        const int c = waveId * 16 + rr;
        const float* Srow = S + ((size_t)bh * DH + c) * DH;
        float qn = fmaxf(sqrtf(nq[c]), 1e-12f);

        float v[4];
#pragma unroll
        for (int t = 0; t < 4; ++t) {
            int d   = lane + 32 * t;
            float kn = fmaxf(sqrtf(nk[d]), 1e-12f);
            v[t] = Srow[d] * temp / (qn * kn);
        }
        float m = fmaxf(fmaxf(v[0], v[1]), fmaxf(v[2], v[3]));
        for (int off = 16; off > 0; off >>= 1)
            m = fmaxf(m, __shfl_xor(m, off));
        float sum = 0.f;
#pragma unroll
        for (int t = 0; t < 4; ++t) {
            v[t] = __expf(v[t] - m);
            sum += v[t];
        }
        for (int off = 16; off > 0; off >>= 1)
            sum += __shfl_xor(sum, off);
        float inv = 1.f / sum;
#pragma unroll
        for (int t = 0; t < 4; ++t) {
            int d = lane + 32 * t;
            attn[((size_t)bh * DH + c) * DH + d] = f2bf(v[t] * inv);
        }
    }
}

// ---------------------------------------------------------------------------
// Kernel 4: fused x_ca = attn @ v  (per head, staged to LDS as [n][C]) and
// out = x_ca^T @ W_out^T + b_out.  Grid B*N/128 = 256; block 256.
// Dynamic LDS: xca 128*512*2 = 128 KB  +  vt 128*136*2 = 34 KB  (WGP has 320 KB).
// ---------------------------------------------------------------------------
__global__ void k_av_out(const unsigned short* __restrict__ qkv,
                         const unsigned short* __restrict__ attn,
                         const unsigned short* __restrict__ wout_b,
                         const float* __restrict__ b_out,
                         float* __restrict__ out) {
    extern __shared__ unsigned short smem[];
    unsigned short* xca = smem;               // [n(128)][cglobal(512)]
    unsigned short* vt  = smem + 128 * 512;   // [n(128)][d(128)], ld=136

    const int tid    = threadIdx.x;
    const int waveId = tid >> 5;
    const int lane   = tid & 31;
    const int b      = blockIdx.x >> 6;              // token tile: 64 per batch
    const int n0     = (blockIdx.x & 63) * 128;

    // ---- phase 1: per-head attn @ v -> xca ----
    for (int h = 0; h < HEADS; ++h) {
        // stage v^T : vt[n][d]
        {
            const unsigned short* vg =
                qkv + ((size_t)(2 * BB + b) * HEADS + h) * (size_t)DH * NN;
            const int d    = tid >> 1;
            const int half = tid & 1;
#pragma unroll
            for (int p = 0; p < 8; ++p) {
                uint4 raw = *(const uint4*)(vg + (size_t)d * NN + n0 +
                                            half * 64 + p * 8);
                const unsigned short* u = (const unsigned short*)&raw;
#pragma unroll
                for (int e = 0; e < 8; ++e)
                    vt[(half * 64 + p * 8 + e) * 136 + d] = u[e];
            }
        }
        __syncthreads();

        const unsigned short* abase = attn + (size_t)(b * HEADS + h) * DH * DH;
        v8f acc[8];
#pragma unroll
        for (int j = 0; j < 8; ++j) acc[j] = vzero();
#pragma unroll
        for (int kk = 0; kk < 4; ++kk) {
            v16bf a = load_frag_a(abase, DH, waveId * 16, kk * 32);
#pragma unroll
            for (int j = 0; j < 8; ++j) {
                v16bf bm = load_frag_b(vt, 136, j * 16, kk * 32);
                acc[j] = WMMA_BF16(a, bm, acc[j]);
            }
        }
        // write into xca[n][h*128 + c]
#pragma unroll
        for (int j = 0; j < 8; ++j)
#pragma unroll
            for (int e = 0; e < 8; ++e) {
                int c  = waveId * 16 + e + ((lane >> 4) << 3);
                int nc = j * 16 + (lane & 15);
                xca[nc * 512 + h * DH + c] = f2bf(acc[j][e]);
            }
        __syncthreads();
    }

    // ---- phase 2: out[n, o] = xca[n, :] . W_out[o, :] + b_out[o] ----
    for (int ob = 0; ob < 4; ++ob) {
        const int o0 = ob * 128;
        v8f acc[8];
#pragma unroll
        for (int j = 0; j < 8; ++j) acc[j] = vzero();
        for (int kk = 0; kk < CC / 32; ++kk) {
            v16bf a = load_frag_a(xca, 512, waveId * 16, kk * 32);
#pragma unroll
            for (int j = 0; j < 8; ++j) {
                v16bf bm = load_frag_b(wout_b, CC, o0 + j * 16, kk * 32);
                acc[j] = WMMA_BF16(a, bm, acc[j]);
            }
        }
#pragma unroll
        for (int j = 0; j < 8; ++j)
#pragma unroll
            for (int e = 0; e < 8; ++e) {
                int n = waveId * 16 + e + ((lane >> 4) << 3);
                int o = o0 + j * 16 + (lane & 15);
                out[((size_t)b * NN + n0 + n) * CC + o] = acc[j][e] + b_out[o];
            }
    }
}

// ---------------------------------------------------------------------------
// Launch
// ---------------------------------------------------------------------------
extern "C" void kernel_launch(void* const* d_in, const int* in_sizes, int n_in,
                              void* d_out, int out_size, void* d_ws, size_t ws_size,
                              hipStream_t stream) {
    const float* x      = (const float*)d_in[0];
    const float* W_qkv  = (const float*)d_in[1];
    const float* W_out  = (const float*)d_in[2];
    const float* b_out  = (const float*)d_in[3];
    const float* temp   = (const float*)d_in[4];
    float* out          = (float*)d_out;

    char* ws = (char*)d_ws;
    size_t off = 0;
    unsigned short* qkv   = (unsigned short*)(ws + off); off += (size_t)3 * BB * HEADS * DH * NN * 2;   // 96 MB
    unsigned short* wqkvb = (unsigned short*)(ws + off); off += (size_t)C3 * CC * 2;                    // 1.5 MB
    unsigned short* woutb = (unsigned short*)(ws + off); off += (size_t)CC * CC * 2;                    // 0.5 MB
    float*          norms = (float*)(ws + off);          off += (size_t)2 * BB * HEADS * DH * 4;        // 16 KB
    float*          S     = (float*)(ws + off);          off += (size_t)BB * HEADS * DH * DH * 4;       // 1 MB
    unsigned short* attn  = (unsigned short*)(ws + off); off += (size_t)BB * HEADS * DH * DH * 2;       // 0.5 MB

    // 0. weights -> bf16
    k_convert_weights<<<(C3 * CC + 255) / 256, 256, 0, stream>>>(W_qkv, W_out, wqkvb, woutb);
    // 1. qkv GEMM (channel-major bf16 out)
    k_qkv_gemm<<<dim3((BB * NN) / 128, C3 / 128), 256, 0, stream>>>(x, wqkvb, qkv);
    // 2. channel L2 norms (sum of squares)
    k_norms<<<2 * BB * HEADS * DH, 256, 0, stream>>>(qkv, norms);
    // 3. S = q k^T  (zero, split-K accumulate, softmax)
    k_zero<<<256, 256, 0, stream>>>(S, BB * HEADS * DH * DH);
    k_qk_gemm<<<dim3(BB * HEADS, 16), 256, 0, stream>>>(qkv, S);
    k_softmax<<<BB * HEADS, 256, 0, stream>>>(S, norms, temp, attn);
    // 4. fused attn@v + output projection
    k_av_out<<<BB * (NN / 128), 256, (128 * 512 + 128 * 136) * 2, stream>>>(
        qkv, attn, woutb, b_out, out);
}
